// GRU_AE_shared_26731876450993
// MI455X (gfx1250) — matrix operand order repaired and checked by reading, not compiled, not run on hardware
//
#include <hip/hip_runtime.h>
#include <hip/hip_bf16.h>

// GRU autoencoder for MI455X (gfx1250), wave32 + WMMA + TDM.
// H=512, T=512, B=128. 16 persistent workgroups, 256 threads (8 waves) each.
// Each WG owns 32 h-columns => 96 rows of Wh (r,z,n gates), staged in LDS as fp16
// once per phase via TENSOR_LOAD_TO_LDS (hardware row padding -> bank-spread
// layout), falling back to a VALU convert loop if workspace is too small for
// the pre-converted fp16 slabs. Per step: h(128x512,f16) @ WhSlice^T(512x96) via
// v_wmma_f32_16x16x32_f16 (8 M-tiles x 6 N-tiles x 16 K-steps, one M-row per wave).
// Recurrent h kept in fp32 registers (each wave owns a disjoint 16x32 tile),
// double-buffered fp16 copy in global scratch for cross-WG A-operand loads.
// Cross-step sync: atomic-counter grid barrier with s_sleep spin.

#define Hh   512
#define Tt   512
#define Bb   128
#define NWG  16
#define RSW  520                      // padded LDS row stride in halfs (bank-spread)
#define SMEM_BYTES (96 * RSW * 2 + Hh * 4)

// workspace layout (bytes)
#define WS_COUNTER_OFF 0
#define WS_HBUF_OFF    256
#define WS_SLAB_ENC_OFF (256 + 2 * Bb * Hh * 2)                 // 262400
#define WS_SLAB_SZ     (NWG * 96 * Hh * 2)                      // 1572864 per matrix
#define WS_SLAB_DEC_OFF (WS_SLAB_ENC_OFF + WS_SLAB_SZ)
#define WS_NEED_TDM    (WS_SLAB_DEC_OFF + WS_SLAB_SZ)

typedef __attribute__((ext_vector_type(16))) _Float16 v16h;
typedef __attribute__((ext_vector_type(8)))  float    v8f;
typedef __attribute__((ext_vector_type(4)))  unsigned int v4u;
typedef __attribute__((ext_vector_type(8)))  int      v8i_t;
typedef __attribute__((ext_vector_type(4)))  int      v4i_t;

union Frag { v16h h; uint4 u[2]; };

struct GateParams { float wi[2][3]; float bi[2][3]; float bh[2][3]; };

__device__ __forceinline__ float sigmoidf(float v) {
  return 1.0f / (1.0f + __expf(-v));
}
__device__ __forceinline__ float tanh_fast(float v) {
  float e = __expf(2.0f * v);
  return 1.0f - 2.0f / (e + 1.0f);   // saturates correctly at +/-1
}

__device__ __forceinline__ void grid_barrier(int* counter, int target, int tid) {
  __threadfence();                   // make this thread's global stores visible (agent)
  __syncthreads();
  if (tid == 0) {
    __hip_atomic_fetch_add(counter, 1, __ATOMIC_ACQ_REL, __HIP_MEMORY_SCOPE_AGENT);
    while (__hip_atomic_load(counter, __ATOMIC_ACQUIRE, __HIP_MEMORY_SCOPE_AGENT) < target) {
      __builtin_amdgcn_s_sleep(1);
    }
  }
  __syncthreads();
}

// ---- TDM: DMA this WG's pre-converted 96x512 fp16 slab into LDS with
// hardware row padding (4 DWORDs after every 256 DWORDs => RSW=520 halfs). ----
__device__ __forceinline__ void tdm_load_slab_to_lds(const _Float16* __restrict__ gsrc,
                                                     _Float16* ldst) {
  // Generic LDS pointer: low 32 bits are the LDS byte address (aperture layout).
  unsigned lds_off = (unsigned)(unsigned long long)(uintptr_t)ldst;
  unsigned long long ga = (unsigned long long)(uintptr_t)gsrc;

  v4u g0;
  g0.x = 1u;                                           // count=1 (valid descriptor)
  g0.y = lds_off;                                      // lds_addr
  g0.z = (unsigned)(ga & 0xFFFFFFFFu);                 // global_addr[31:0]
  g0.w = (unsigned)((ga >> 32) & 0x01FFFFFFu)          // global_addr[56:32]
       | (2u << 30);                                   // type=2 ("image")

  v8i_t g1;
  g1[0] = (int)((2u << 16)        // data_size = 4 bytes (dims in DWORDs)
              | (1u << 20)        // pad_enable
              | (7u << 22)        // pad_interval: 256 DWORDs (one 512-half row)
              | (3u << 25));      // pad_amount: 4 DWORDs (16B -> RSW=520)
  g1[1] = (int)(256u << 16);      // tensor_dim0[15:0]=256 (row length, DWORDs)
  g1[2] = (int)(96u << 16);       // tensor_dim0 hi=0 | tensor_dim1[15:0]=96 rows
  g1[3] = (int)(256u << 16);      // tensor_dim1 hi=0 | tile_dim0=256
  g1[4] = (int)96u;               // tile_dim1=96 | tile_dim2=0
  g1[5] = (int)256u;              // tensor_dim0_stride[31:0]=256 (contiguous rows)
  g1[6] = 0;                      // stride0 hi | stride1 lo
  g1[7] = 0;

  v4i_t z4 = {0, 0, 0, 0};
#if __has_include(<hip/amd_detail/amd_gfx1250_TDM.h>)
  v8i_t z8 = {0, 0, 0, 0, 0, 0, 0, 0};
  __builtin_amdgcn_tensor_load_to_lds(g0, g1, z4, z4, z8, 0);
#else
  __builtin_amdgcn_tensor_load_to_lds(g0, g1, z4, z4, 0);
#endif
}

__device__ __forceinline__ void load_weights_tdm(const _Float16* __restrict__ slab_wg,
                                                 const float* __restrict__ lW,
                                                 _Float16* wlds, float* linw, int tid) {
  if (tid < 32) {                        // one wave issues the DMA
    tdm_load_slab_to_lds(slab_wg, wlds);
    __builtin_amdgcn_s_wait_tensorcnt(0);
  }
  for (int k = tid; k < Hh; k += 256) linw[k] = lW[k];
  __syncthreads();
}

// Fallback: stage this WG's 96 Wh rows (fp32 global -> fp16 LDS) + lin_W.
__device__ __forceinline__ void load_weights_convert(const float* __restrict__ Wh,
                                                     const float* __restrict__ lW,
                                                     _Float16* wlds, float* linw,
                                                     int wg, int tid) {
  for (int idx = tid; idx < 96 * Hh; idx += 256) {
    int jl = idx >> 9, k = idx & (Hh - 1);
    int j  = (jl >> 5) * Hh + wg * 32 + (jl & 31);   // gate-major Wh row
    wlds[jl * RSW + k] = (_Float16)Wh[j * Hh + k];
  }
  for (int k = tid; k < Hh; k += 256) linw[k] = lW[k];
  __syncthreads();
}

__device__ __forceinline__ void load_gate_params(const float* __restrict__ Wi,
                                                 const float* __restrict__ bi,
                                                 const float* __restrict__ bh,
                                                 int wg, int ln, GateParams& p) {
#pragma unroll
  for (int nt = 0; nt < 2; ++nt)
#pragma unroll
    for (int g = 0; g < 3; ++g) {
      int j = g * Hh + wg * 32 + nt * 16 + ln;
      p.wi[nt][g] = Wi[j];
      p.bi[nt][g] = bi[j];
      p.bh[nt][g] = bh[j];
    }
}

// acc[n] = h_tile(16x512) @ WhSlice^T tile(512x16n). A from global fp16 h,
// B from LDS weight slab. A and B both use contiguous k = 16*hi + s per lane (not the ISA table order;
// same order in A and B, so valid); f32 C/D 16x16.
__device__ __forceinline__ void matmul96(const _Float16* __restrict__ hsrc,
                                         const _Float16* __restrict__ wlds,
                                         int wave, int lane, v8f acc[6]) {
  const int ln = lane & 15;
  const int hi = (lane >> 4) & 1;
  const _Float16* arow = hsrc + (16 * wave + ln) * Hh + 16 * hi;
  v8f zero = {0.f, 0.f, 0.f, 0.f, 0.f, 0.f, 0.f, 0.f};
#pragma unroll
  for (int n = 0; n < 6; ++n) acc[n] = zero;
#pragma unroll 2
  for (int kk = 0; kk < 16; ++kk) {
    const int k0 = kk * 32;
    Frag a;
    const uint4* ap = (const uint4*)(arow + k0);
    a.u[0] = ap[0];            // K = k0 + 16*hi + [0..7]
    a.u[1] = ap[1];            // K = k0 + 16*hi + 8 + [0..7]
#pragma unroll
    for (int n = 0; n < 6; ++n) {
      Frag b;
      const uint4* bp = (const uint4*)(wlds + (n * 16 + ln) * RSW + k0 + 16 * hi);
      b.u[0] = bp[0];          // 16 contiguous K values for this lane's column
      b.u[1] = bp[1];
      acc[n] = __builtin_amdgcn_wmma_f32_16x16x32_f16(
          false, a.h, false, b.h, (short)0, acc[n], false, false);
    }
  }
}

// Per-lane gate math + fp32 register h update + fp16 store of owned tile.
__device__ __forceinline__ void gate_update(const v8f acc[6], const GateParams& p,
                                            const float inb[8], float hreg[2][8],
                                            _Float16* __restrict__ hdst,
                                            int wave, int lane, int wg) {
  const int ln = lane & 15;
  const int hi = (lane >> 4) & 1;
#pragma unroll
  for (int nt = 0; nt < 2; ++nt) {
    const int hc = wg * 32 + nt * 16 + ln;
#pragma unroll
    for (int e = 0; e < 8; ++e) {
      const int b = 16 * wave + 8 * hi + e;
      float xin = inb[e];
      float r = sigmoidf(xin * p.wi[nt][0] + p.bi[nt][0] + acc[nt][e]     + p.bh[nt][0]);
      float z = sigmoidf(xin * p.wi[nt][1] + p.bi[nt][1] + acc[nt + 2][e] + p.bh[nt][1]);
      float gn =         xin * p.wi[nt][2] + p.bi[nt][2] + r * (acc[nt + 4][e] + p.bh[nt][2]);
      float n  = tanh_fast(gn);
      float hn = (1.0f - z) * n + z * hreg[nt][e];
      hreg[nt][e] = hn;
      hdst[b * Hh + hc] = (_Float16)hn;
    }
  }
}

// y[b] = h[b,:] . lin_W  (per-wave: lanes 0..15 -> rows, halves split K).
__device__ __forceinline__ float compute_y(const _Float16* __restrict__ hsrc,
                                           const float* __restrict__ linw,
                                           int wave, int lane) {
  const int bq = 16 * wave + (lane & 15);
  const int k0 = (lane & 16) ? 256 : 0;
  const _Float16* hp = hsrc + bq * Hh + k0;
  const float* wp = linw + k0;
  float s = 0.0f;
#pragma unroll 4
  for (int k = 0; k < 256; k += 8) {
    Frag a;
    a.u[0] = *(const uint4*)(hp + k);
#pragma unroll
    for (int i = 0; i < 8; ++i) s += (float)a.h[i] * wp[k + i];
  }
  s += __shfl_xor(s, 16, 32);
  return s;
}

__global__ void gru_ae_init_kernel(char* __restrict__ ws,
                                   const float* __restrict__ encWh,
                                   const float* __restrict__ decWh,
                                   int make_slabs) {
  int idx    = blockIdx.x * blockDim.x + threadIdx.x;
  int stride = gridDim.x * blockDim.x;
  if (idx < 64) ((int*)ws)[idx] = 0;                   // barrier counter area
  _Float16* h0 = (_Float16*)(ws + WS_HBUF_OFF);
  for (int i = idx; i < Bb * Hh; i += stride)
    h0[i] = (_Float16)0.0f;                            // h_0 = 0 (buffer 0)
  if (make_slabs) {
    // Pre-convert Wh to fp16, gate-major, one contiguous 96x512 slab per WG,
    // so the main kernel can TDM them straight into LDS.
    _Float16* encS = (_Float16*)(ws + WS_SLAB_ENC_OFF);
    _Float16* decS = (_Float16*)(ws + WS_SLAB_DEC_OFF);
    for (int i = idx; i < NWG * 96 * Hh; i += stride) {
      int wg = i / (96 * Hh);
      int r  = i - wg * 96 * Hh;
      int jl = r >> 9, k = r & (Hh - 1);
      int j  = (jl >> 5) * Hh + wg * 32 + (jl & 31);
      encS[i] = (_Float16)encWh[j * Hh + k];
      decS[i] = (_Float16)decWh[j * Hh + k];
    }
  }
}

__global__ void __launch_bounds__(256, 1)
gru_ae_main_kernel(const float* __restrict__ x,
                   const float* __restrict__ enc_Wi, const float* __restrict__ enc_Wh,
                   const float* __restrict__ enc_bi, const float* __restrict__ enc_bh,
                   const float* __restrict__ dec_Wi, const float* __restrict__ dec_Wh,
                   const float* __restrict__ dec_bi, const float* __restrict__ dec_bh,
                   const float* __restrict__ lin_W, const float* __restrict__ lin_b,
                   float* __restrict__ out, char* __restrict__ ws, int use_tdm) {
  extern __shared__ char smem[];
  _Float16* wlds = (_Float16*)smem;
  float* linw = (float*)(smem + 96 * RSW * 2);

  const int tid  = threadIdx.x;
  const int wg   = blockIdx.x;
  const int lane = tid & 31;
  const int wave = tid >> 5;
  const int ln   = lane & 15;
  const int hi   = (lane >> 4) & 1;

  int* counter = (int*)ws;
  _Float16* hbuf0 = (_Float16*)(ws + WS_HBUF_OFF);
  _Float16* hbuf1 = hbuf0 + Bb * Hh;
  const _Float16* enc_slab = (const _Float16*)(ws + WS_SLAB_ENC_OFF) + wg * 96 * Hh;
  const _Float16* dec_slab = (const _Float16*)(ws + WS_SLAB_DEC_OFF) + wg * 96 * Hh;

  int epoch = 0;
  int s = 0;
  float hreg[2][8];
#pragma unroll
  for (int nt = 0; nt < 2; ++nt)
#pragma unroll
    for (int e = 0; e < 8; ++e) hreg[nt][e] = 0.0f;

  GateParams p;

  // ---------------- encoder ----------------
  if (use_tdm) load_weights_tdm(enc_slab, lin_W, wlds, linw, tid);
  else         load_weights_convert(enc_Wh, lin_W, wlds, linw, wg, tid);
  load_gate_params(enc_Wi, enc_bi, enc_bh, wg, ln, p);

#pragma unroll 1
  for (int t = 0; t < Tt; ++t, ++s) {
    const _Float16* hcur = (s & 1) ? hbuf1 : hbuf0;
    _Float16*       hnxt = (s & 1) ? hbuf0 : hbuf1;
    float xv = (lane < 16) ? x[(16 * wave + lane) * Tt + t] : 0.0f;
    v8f acc[6];
    matmul96(hcur, wlds, wave, lane, acc);
    float inb[8];
#pragma unroll
    for (int e = 0; e < 8; ++e) inb[e] = __shfl(xv, e + 8 * hi, 32);
    gate_update(acc, p, inb, hreg, hnxt, wave, lane, wg);
    grid_barrier(counter, NWG * (++epoch), tid);
  }

  // ---------------- decoder ----------------
  if (use_tdm) load_weights_tdm(dec_slab, lin_W, wlds, linw, tid);
  else         load_weights_convert(dec_Wh, lin_W, wlds, linw, wg, tid);
  load_gate_params(dec_Wi, dec_bi, dec_bh, wg, ln, p);
  const float lb = lin_b[0];

#pragma unroll 1
  for (int t = 0; t < Tt; ++t, ++s) {
    const _Float16* hcur = (s & 1) ? hbuf1 : hbuf0;
    _Float16*       hnxt = (s & 1) ? hbuf0 : hbuf1;
    float yv = compute_y(hcur, linw, wave, lane) + lb;   // y_t from h_t
    if (t > 0 && wg == 0 && lane < 16)
      out[(16 * wave + lane) * Tt + (Tt - t)] = yv;      // ys[t-1] flipped
    v8f acc[6];
    matmul96(hcur, wlds, wave, lane, acc);
    float inb[8];
#pragma unroll
    for (int e = 0; e < 8; ++e) inb[e] = __shfl(yv, e + 8 * hi, 32);
    gate_update(acc, p, inb, hreg, hnxt, wave, lane, wg);
    grid_barrier(counter, NWG * (++epoch), tid);
  }

  // final emission: ys[T-1] = y from h_T  -> out[:, 0]
  {
    const _Float16* hcur = (s & 1) ? hbuf1 : hbuf0;
    float yv = compute_y(hcur, linw, wave, lane) + lb;
    if (wg == 0 && lane < 16)
      out[(16 * wave + lane) * Tt + 0] = yv;
  }
}

extern "C" void kernel_launch(void* const* d_in, const int* in_sizes, int n_in,
                              void* d_out, int out_size, void* d_ws, size_t ws_size,
                              hipStream_t stream) {
  (void)in_sizes; (void)n_in; (void)out_size;
  const float* x      = (const float*)d_in[0];
  const float* enc_Wi = (const float*)d_in[1];
  const float* enc_Wh = (const float*)d_in[2];
  const float* enc_bi = (const float*)d_in[3];
  const float* enc_bh = (const float*)d_in[4];
  const float* dec_Wi = (const float*)d_in[5];
  const float* dec_Wh = (const float*)d_in[6];
  const float* dec_bi = (const float*)d_in[7];
  const float* dec_bh = (const float*)d_in[8];
  const float* lin_W  = (const float*)d_in[9];
  const float* lin_b  = (const float*)d_in[10];
  float* out = (float*)d_out;
  char*  ws  = (char*)d_ws;

  const int use_tdm = (ws_size >= (size_t)WS_NEED_TDM) ? 1 : 0;

  (void)hipFuncSetAttribute((const void*)gru_ae_main_kernel,
                            hipFuncAttributeMaxDynamicSharedMemorySize, SMEM_BYTES);

  gru_ae_init_kernel<<<dim3(512), dim3(256), 0, stream>>>(ws, enc_Wh, dec_Wh, use_tdm);
  gru_ae_main_kernel<<<dim3(NWG), dim3(256), SMEM_BYTES, stream>>>(
      x, enc_Wi, enc_Wh, enc_bi, enc_bh,
      dec_Wi, dec_Wh, dec_bi, dec_bh, lin_W, lin_b, out, ws, use_tdm);
}
